// Baseline_20761871909173
// MI455X (gfx1250) — matrix operand order repaired and checked by reading, not compiled, not run on hardware
//
#include <hip/hip_runtime.h>
#include <hip/hip_bf16.h>
#include <math.h>

// ---------------------------------------------------------------------------
// Problem constants (from reference):  B=128, S=512, EMB=512, HID=1024, OUT=1
// ---------------------------------------------------------------------------
#define Bk 128
#define Sk 512
#define Ek 512
#define Hk 1024

typedef __bf16 bf16;
typedef __attribute__((ext_vector_type(16))) __bf16 v16bf;
typedef __attribute__((ext_vector_type(8)))  float  v8f;
typedef __attribute__((ext_vector_type(4)))  unsigned int v4u;

union ABf { v16bf v; v4u u[2]; };

// A-operand (16x32 bf16, MxK): lane holds k%32 in {off..off+7} and
// {off+16..off+23}, off = (lane>=16)*8  -> two 16B runs, second at +16 elems.
static __device__ __forceinline__ v16bf load_a(const bf16* p) {
  ABf t;
  t.u[0] = *(const v4u*)(p);
  t.u[1] = *(const v4u*)(p + 16);
  return t.v;
}
// B-operand (32x16 bf16, KxN): lane (column n) holds k%32 in {off..off+7} and
// {off+16..off+23}, off = (lane>=16)*8 -> two 16B runs, second at +16 elems.
static __device__ __forceinline__ v16bf load_b(const bf16* p) {
  ABf t;
  t.u[0] = *(const v4u*)(p);
  t.u[1] = *(const v4u*)(p + 16);
  return t.v;
}

static __device__ __forceinline__ v8f wmma_bf16(v16bf a, v16bf b, v8f c) {
  // D = A*B + C, fp32 accumulate
  return __builtin_amdgcn_wmma_f32_16x16x32_bf16(false, a, false, b,
                                                 (short)0, c, false, false);
}

// ---------------------------------------------------------------------------
// fp32 -> bf16 hi/lo split (hi = rn(x), lo = rn(x - hi)); hi*hi+hi*lo+lo*hi
// recovers ~fp32 accuracy on the bf16 matrix pipe at fp32 memory traffic.
// ---------------------------------------------------------------------------
__global__ void pack_hilo_kernel(const float* __restrict__ src,
                                 bf16* __restrict__ hi, bf16* __restrict__ lo,
                                 size_t n) {
  size_t i = (size_t)blockIdx.x * blockDim.x + threadIdx.x;
  if (i < n) {
    float v = src[i];
    bf16 h = (bf16)v;
    hi[i] = h;
    lo[i] = (bf16)(v - (float)h);
  }
}

// ---------------------------------------------------------------------------
// Phase 1: pre[s][b][h] = emb[x[b][s]] @ W^T + W_b
// One block per 16-row m-tile (m = s*128 + b). 8 waves x 8 n-tiles each.
// ---------------------------------------------------------------------------
__global__ __launch_bounds__(256) void phase1_kernel(
    const int* __restrict__ x, const float* __restrict__ emb,
    const bf16* __restrict__ Whi, const bf16* __restrict__ Wlo,
    const float* __restrict__ Wb, float* __restrict__ pre) {
  __shared__ __align__(16) bf16 sAhi[16][Ek];
  __shared__ __align__(16) bf16 sAlo[16][Ek];

  const unsigned mt = blockIdx.x;       // 0..4095
  const unsigned s  = mt >> 3;
  const unsigned bt = mt & 7;
  const unsigned tid = threadIdx.x;

  // Gather 16 embedding rows, split into bf16 hi/lo in LDS.
  {
    const unsigned row = tid >> 4;            // 0..15
    const unsigned c0  = (tid & 15) * 32;     // 16 chunks of 32 floats
    const unsigned b   = bt * 16 + row;
    const int tok = x[(size_t)b * Sk + s];
    const float* er = emb + (size_t)tok * Ek + c0;
#pragma unroll
    for (int i = 0; i < 32; ++i) {
      float v = er[i];
      bf16 h = (bf16)v;
      sAhi[row][c0 + i] = h;
      sAlo[row][c0 + i] = (bf16)(v - (float)h);
    }
  }
  __syncthreads();

  const unsigned lane = tid & 31;
  const unsigned wv   = tid >> 5;             // 0..7
  const unsigned hi16 = (lane >> 4) & 1;
  const unsigned l15  = lane & 15;

  for (unsigned nti = 0; nti < 8; ++nti) {
    const unsigned nt = wv * 8 + nti;         // 0..63
    const unsigned n  = nt * 16 + l15;
    v8f acc = {};
    for (unsigned kt = 0; kt < Ek / 32; ++kt) {
      v16bf ah = load_a(&sAhi[l15][kt * 32 + hi16 * 8]);
      v16bf al = load_a(&sAlo[l15][kt * 32 + hi16 * 8]);
      v16bf bh = load_b(Whi + (size_t)n * Ek + kt * 32 + hi16 * 8);
      v16bf bl = load_b(Wlo + (size_t)n * Ek + kt * 32 + hi16 * 8);
      acc = wmma_bf16(ah, bh, acc);
      acc = wmma_bf16(ah, bl, acc);
      acc = wmma_bf16(al, bh, acc);
    }
    const float bias = Wb[n];
#pragma unroll
    for (int r = 0; r < 8; ++r) {             // D layout: m = r + 8*(lane/16)
      unsigned b = bt * 16 + r + hi16 * 8;
      pre[((size_t)s * Bk + b) * Hk + n] = acc[r] + bias;
    }
  }
}

// ---------------------------------------------------------------------------
// Group-local barrier (8 WGs per batch group; cnt/gen zeroed each launch).
// ---------------------------------------------------------------------------
static __device__ __forceinline__ void group_sync(unsigned* cnt, unsigned* gen,
                                                  unsigned nblocks,
                                                  unsigned target) {
  __syncthreads();
  if (threadIdx.x == 0) {
    __threadfence();
    unsigned prev = __hip_atomic_fetch_add(cnt, 1u, __ATOMIC_ACQ_REL,
                                           __HIP_MEMORY_SCOPE_AGENT);
    if (prev == nblocks - 1) {
      __hip_atomic_store(cnt, 0u, __ATOMIC_RELAXED, __HIP_MEMORY_SCOPE_AGENT);
      __hip_atomic_fetch_add(gen, 1u, __ATOMIC_ACQ_REL,
                             __HIP_MEMORY_SCOPE_AGENT);
    }
    while (__hip_atomic_load(gen, __ATOMIC_ACQUIRE,
                             __HIP_MEMORY_SCOPE_AGENT) < target) {
      __builtin_amdgcn_s_sleep(2);
    }
  }
  __syncthreads();
}

// ---------------------------------------------------------------------------
// Phase 2: persistent cooperative recurrence.
// 64 blocks x 8 waves = 512 waves = one 16x16 tile of h[128][1024] per wave.
// WGs decompose into 8 *independent* groups of 8 (one per 16-row batch tile);
// each group syncs only with itself. The group's 64KB h slice (bf16 hi/lo)
// is staged in LDS once per step (8x traffic reduction on the A operand).
// ---------------------------------------------------------------------------
__global__ __launch_bounds__(256) void rnn_kernel(
    const bf16* __restrict__ Uhi, const bf16* __restrict__ Ulo,
    const float* __restrict__ pre, const float* __restrict__ Ub,
    bf16* h_hi0, bf16* h_lo0, bf16* h_hi1, bf16* h_lo1,
    float* __restrict__ hT, unsigned* bar_base) {
  __shared__ __align__(16) bf16 sHhi[16 * Hk];   // 32 KB
  __shared__ __align__(16) bf16 sHlo[16 * Hk];   // 32 KB

  const unsigned g    = blockIdx.x;          // 0..63
  const unsigned tid  = threadIdx.x;
  const unsigned lane = tid & 31;
  const unsigned wv   = tid >> 5;            // 0..7
  const unsigned mt   = g >> 3;              // batch tile / barrier group 0..7
  const unsigned nt   = (g & 7) * 8 + wv;    // hidden tile 0..63
  const unsigned hi16 = (lane >> 4) & 1;
  const unsigned l15  = lane & 15;
  const unsigned n    = nt * 16 + l15;       // output hidden index
  const float ubn = Ub[n];

  unsigned* bcnt = bar_base + (size_t)mt * 32;   // 128B-separated per group
  unsigned* bgen = bcnt + 1;

  bf16* bufs_hi[2] = {h_hi0, h_hi1};
  bf16* bufs_lo[2] = {h_lo0, h_lo1};

  for (unsigned t = 0; t < Sk; ++t) {
    const bf16* cur_hi = bufs_hi[t & 1];
    const bf16* cur_lo = bufs_lo[t & 1];
    bf16* nxt_hi = bufs_hi[(t + 1) & 1];
    bf16* nxt_lo = bufs_lo[(t + 1) & 1];

    // Stage this group's 16 h rows (contiguous 32KB hi + 32KB lo) into LDS.
    {
      const v4u* srcH = (const v4u*)(cur_hi + (size_t)mt * 16 * Hk);
      const v4u* srcL = (const v4u*)(cur_lo + (size_t)mt * 16 * Hk);
      v4u* dstH = (v4u*)sHhi;
      v4u* dstL = (v4u*)sHlo;
#pragma unroll
      for (int i = 0; i < 8; ++i) {          // 2048 uint4 per array
        dstH[tid + i * 256] = srcH[tid + i * 256];
        dstL[tid + i * 256] = srcL[tid + i * 256];
      }
    }
    __syncthreads();

    v8f acc = {};
    for (unsigned kt = 0; kt < Hk / 32; ++kt) {
      v16bf ah = load_a(sHhi + (size_t)l15 * Hk + kt * 32 + hi16 * 8);
      v16bf al = load_a(sHlo + (size_t)l15 * Hk + kt * 32 + hi16 * 8);
      v16bf bh = load_b(Uhi + (size_t)n * Hk + kt * 32 + hi16 * 8);
      v16bf bl = load_b(Ulo + (size_t)n * Hk + kt * 32 + hi16 * 8);
      acc = wmma_bf16(ah, bh, acc);
      acc = wmma_bf16(ah, bl, acc);
      acc = wmma_bf16(al, bh, acc);
    }

    const float* pre_t = pre + (size_t)t * Bk * Hk;
#pragma unroll
    for (int r = 0; r < 8; ++r) {            // D layout: m = r + 8*(lane/16)
      const unsigned b = mt * 16 + r + hi16 * 8;
      float v = acc[r] + pre_t[(size_t)b * Hk + n] + ubn;
      float h = tanhf(v);
      bf16 hh = (bf16)h;
      nxt_hi[(size_t)b * Hk + n] = hh;
      nxt_lo[(size_t)b * Hk + n] = (bf16)(h - (float)hh);
      if (t == Sk - 1) hT[(size_t)b * Hk + n] = h;
    }
    group_sync(bcnt, bgen, 8, t + 1);        // sync only the 8 WGs of this mt
  }
}

// ---------------------------------------------------------------------------
// Phase 3: logits = sigmoid(hT @ V^T + V_b). One wave per batch row,
// coalesced lane-parallel reduction.
// ---------------------------------------------------------------------------
__global__ __launch_bounds__(256) void logits_kernel(
    const float* __restrict__ hT, const float* __restrict__ Vw,
    const float* __restrict__ Vb, float* __restrict__ out) {
  const unsigned lane = threadIdx.x & 31;
  const unsigned wv   = threadIdx.x >> 5;
  const unsigned b    = blockIdx.x * 8 + wv;   // grid 16 x 8 waves = 128 rows
  float s = 0.f;
  for (unsigned k = lane; k < Hk; k += 32)
    s += hT[(size_t)b * Hk + k] * Vw[k];
#pragma unroll
  for (int off = 16; off > 0; off >>= 1)
    s += __shfl_xor(s, off, 32);
  if (lane == 0)
    out[b] = 1.f / (1.f + expf(-(s + Vb[0])));
}

// ---------------------------------------------------------------------------
// Launch
// ---------------------------------------------------------------------------
extern "C" void kernel_launch(void* const* d_in, const int* in_sizes, int n_in,
                              void* d_out, int out_size, void* d_ws,
                              size_t ws_size, hipStream_t stream) {
  (void)in_sizes; (void)n_in; (void)out_size; (void)ws_size;
  const int*   x   = (const int*)d_in[0];
  const float* emb = (const float*)d_in[1];
  const float* Ww  = (const float*)d_in[2];
  const float* Wb  = (const float*)d_in[3];
  const float* Uw  = (const float*)d_in[4];
  const float* Ub  = (const float*)d_in[5];
  const float* Vw  = (const float*)d_in[6];
  const float* Vb  = (const float*)d_in[7];
  float* out = (float*)d_out;
  char*  ws  = (char*)d_ws;

  // Workspace layout (bytes)
  constexpr size_t BAR = 0;                                   // 1 KB (8 grps)
  constexpr size_t UHI = 1024;                                // 2 MB
  constexpr size_t ULO = UHI + (size_t)Hk * Hk * 2;           // 2 MB
  constexpr size_t WHI = ULO + (size_t)Hk * Hk * 2;           // 1 MB
  constexpr size_t WLO = WHI + (size_t)Hk * Ek * 2;           // 1 MB
  constexpr size_t H0H = WLO + (size_t)Hk * Ek * 2;           // 256 KB
  constexpr size_t H0L = H0H + (size_t)Bk * Hk * 2;
  constexpr size_t H1H = H0L + (size_t)Bk * Hk * 2;
  constexpr size_t H1L = H1H + (size_t)Bk * Hk * 2;
  constexpr size_t HT  = H1L + (size_t)Bk * Hk * 2;           // 512 KB
  constexpr size_t PRE = HT + (size_t)Bk * Hk * 4;            // 256 MiB

  // Deterministic per-call init: barriers + h0 = 0
  hipMemsetAsync(ws + BAR, 0, 1024, stream);
  hipMemsetAsync(ws + H0H, 0, (size_t)Bk * Hk * 2 * 2, stream);

  pack_hilo_kernel<<<(Hk * Hk + 255) / 256, 256, 0, stream>>>(
      Uw, (bf16*)(ws + UHI), (bf16*)(ws + ULO), (size_t)Hk * Hk);
  pack_hilo_kernel<<<(Hk * Ek + 255) / 256, 256, 0, stream>>>(
      Ww, (bf16*)(ws + WHI), (bf16*)(ws + WLO), (size_t)Hk * Ek);

  phase1_kernel<<<(Sk * Bk) / 16, 256, 0, stream>>>(
      x, emb, (const bf16*)(ws + WHI), (const bf16*)(ws + WLO), Wb,
      (float*)(ws + PRE));

  rnn_kernel<<<64, 256, 0, stream>>>(
      (const bf16*)(ws + UHI), (const bf16*)(ws + ULO),
      (const float*)(ws + PRE), Ub,
      (bf16*)(ws + H0H), (bf16*)(ws + H0L),
      (bf16*)(ws + H1H), (bf16*)(ws + H1L),
      (float*)(ws + HT), (unsigned*)(ws + BAR));

  logits_kernel<<<16, 256, 0, stream>>>((const float*)(ws + HT), Vw, Vb, out);
}
